// MergeAttentionSubBlockFull_49675591746166
// MI455X (gfx1250) — matrix operand order repaired and checked by reading, not compiled, not run on hardware
//
#include <hip/hip_runtime.h>
#include <stdint.h>

// ---------------------------------------------------------------------------
// MergeAttentionSubBlockFull for MI455X (gfx1250, wave32, WMMA bf16)
//
//   1) LayerNorm(x) -> xn (bf16)
//   2) weight folding (one-time):
//        UWt_m = U_m @ W_m^T ; W_fold = UWt @ M_qkv ; b_fold = bias @ M_qkv
//        W_ofold_m = U_m @ Wout_m^T   (out-proj folded => single final GEMM)
//      All GEMM B-operands pre-transposed to [N,K] so LDS staging is a pure
//      contiguous copy -> GLOBAL_LOAD_ASYNC_TO_LDS_B128 (ASYNCcnt).
//   3) qkv = xn @ W_fold + b_fold  (bf16)
//   4) flash attention per (b,h): WMMA Q@K^T and P@V, online softmax
//   5) out = o @ W_ofold + out_bias  (fp32 -> d_out)
// ---------------------------------------------------------------------------

typedef __attribute__((ext_vector_type(8)))  float          v8f;
typedef __attribute__((ext_vector_type(16))) __bf16         v16bf;
typedef __attribute__((ext_vector_type(8)))  unsigned short u16x8;
typedef __attribute__((ext_vector_type(16))) unsigned short u16x16;

#define NBATCH 64
#define NT     256
#define NE     768
#define NM     3
#define NH     12
#define HD     64
#define THREE_E 2304   /* 3*EMB   */
#define MK      6912   /* NM*3*EMB */

static __device__ __forceinline__ unsigned short f2bfu(float f) {
  unsigned u = __builtin_bit_cast(unsigned, f);
  u += 0x7FFFu + ((u >> 16) & 1u);        // round-to-nearest-even
  return (unsigned short)(u >> 16);
}

static __device__ __forceinline__ v16bf pack16(u16x8 lo, u16x8 hi) {
  u16x16 t;
#pragma unroll
  for (int i = 0; i < 8; ++i) { t[i] = lo[i]; t[8 + i] = hi[i]; }
  return __builtin_bit_cast(v16bf, t);
}

static __device__ __forceinline__ v8f wmma_bf16(v16bf a, v16bf b, v8f c) {
  return __builtin_amdgcn_wmma_f32_16x16x32_bf16(false, a, false, b,
                                                 (short)0, c, false, false);
}

// --- CDNA5 async global->LDS copy (16B), tracked by ASYNCcnt ----------------
typedef int v4i __attribute__((vector_size(16)));
typedef __attribute__((address_space(1))) v4i* gas_v4i_p;
typedef __attribute__((address_space(3))) v4i* las_v4i_p;

static __device__ __forceinline__ void async_ld16(const unsigned short* gsrc,
                                                  unsigned short* ldst) {
#if __has_builtin(__builtin_amdgcn_global_load_async_to_lds_b128)
  __builtin_amdgcn_global_load_async_to_lds_b128(
      (gas_v4i_p)gsrc, (las_v4i_p)ldst, 0, 0);
#else
  unsigned loff = (unsigned)(uintptr_t)(__attribute__((address_space(3))) void*)ldst;
  asm volatile("global_load_async_to_lds_b128 %0, %1, off"
               :: "v"(loff), "v"(gsrc) : "memory");
#endif
}

static __device__ __forceinline__ void wait_async0() {
#if __has_builtin(__builtin_amdgcn_s_wait_asynccnt)
  __builtin_amdgcn_s_wait_asynccnt(0);
#else
  asm volatile("s_wait_asynccnt 0x0" ::: "memory");
#endif
}

// ---------------------------------------------------------------------------
// fp32 -> bf16 convert (grid-stride)
// ---------------------------------------------------------------------------
__global__ void __launch_bounds__(256)
cvt_f32_bf16(const float* __restrict__ s, unsigned short* __restrict__ d, long n) {
  long i  = (long)blockIdx.x * blockDim.x + threadIdx.x;
  long st = (long)gridDim.x * blockDim.x;
  for (; i < n; i += st) d[i] = f2bfu(s[i]);
}

// ---------------------------------------------------------------------------
// Tiled transpose: dst[c*R + r] = bf16(src[r*C + c]); src fp32 or bf16.
// grid = (C/32, R/32), 256 threads.
// ---------------------------------------------------------------------------
__global__ void __launch_bounds__(256)
transpose_bf16(const void* __restrict__ src, int srcF32,
               unsigned short* __restrict__ dst, int R, int C) {
  __shared__ unsigned short t[32][34];
  const int tx = threadIdx.x & 31, ty = threadIdx.x >> 5;
#pragma unroll
  for (int i = 0; i < 4; ++i) {
    const int r = blockIdx.y * 32 + ty + i * 8, c = blockIdx.x * 32 + tx;
    t[ty + i * 8][tx] = srcF32
        ? f2bfu(((const float*)src)[(size_t)r * C + c])
        : ((const unsigned short*)src)[(size_t)r * C + c];
  }
  __syncthreads();
#pragma unroll
  for (int i = 0; i < 4; ++i) {
    const int c = blockIdx.x * 32 + ty + i * 8, r = blockIdx.y * 32 + tx;
    dst[(size_t)c * R + r] = t[tx][ty + i * 8];
  }
}

// ---------------------------------------------------------------------------
// LayerNorm over last dim (768), one block per row, output bf16
// ---------------------------------------------------------------------------
__global__ void __launch_bounds__(256)
ln_bf16_kernel(const float* __restrict__ x, const float* __restrict__ g,
               const float* __restrict__ be, unsigned short* __restrict__ xn) {
  __shared__ float red[256];
  const int row = blockIdx.x, tid = threadIdx.x;
  const float* xr = x + (size_t)row * NE;
  float v[3], s = 0.f;
#pragma unroll
  for (int i = 0; i < 3; ++i) { v[i] = xr[tid + i * 256]; s += v[i]; }
  red[tid] = s; __syncthreads();
  for (int o = 128; o > 0; o >>= 1) { if (tid < o) red[tid] += red[tid + o]; __syncthreads(); }
  const float mean = red[0] * (1.0f / NE);
  __syncthreads();
  float s2 = 0.f;
#pragma unroll
  for (int i = 0; i < 3; ++i) { float d = v[i] - mean; s2 += d * d; }
  red[tid] = s2; __syncthreads();
  for (int o = 128; o > 0; o >>= 1) { if (tid < o) red[tid] += red[tid + o]; __syncthreads(); }
  const float inv = rsqrtf(red[0] * (1.0f / NE) + 1e-5f);
  unsigned short* orow = xn + (size_t)row * NE;
#pragma unroll
  for (int i = 0; i < 3; ++i) {
    int c = tid + i * 256;
    orow[c] = f2bfu((v[i] - mean) * inv * g[c] + be[c]);
  }
}

// ---------------------------------------------------------------------------
// b_fold[c] = sum_j bias_flat[j] * M_qkv[j,c]
// ---------------------------------------------------------------------------
__global__ void __launch_bounds__(256)
bfold_kernel(const float* __restrict__ bias, const float* __restrict__ M,
             float* __restrict__ out) {
  int c = blockIdx.x * 256 + threadIdx.x;
  if (c >= THREE_E) return;
  float s = 0.f;
  for (int j = 0; j < MK; ++j) s += bias[j] * M[(size_t)j * THREE_E + c];
  out[c] = s;
}

// ---------------------------------------------------------------------------
// bf16 WMMA GEMM: C[M,N] = A[M,K] * B^T + bias, with B stored [N,K] row-major.
// Block tile 128x128x32, 8 waves (2x4), wave tile 64x32 = 8 C-frags.
// Double-buffered staging via GLOBAL_LOAD_ASYNC_TO_LDS_B128 (pure copies).
// ---------------------------------------------------------------------------
#define BM 128
#define BN 128
#define BK 32
#define LP 40   /* padded halfword row stride (80B, 16B aligned) */

__global__ void __launch_bounds__(256)
gemm_bf16_wmma(const unsigned short* __restrict__ A, int lda, long sA,
               const unsigned short* __restrict__ B, int ldb, long sB,
               const float* __restrict__ bias,
               float* __restrict__ outF, unsigned short* __restrict__ outB,
               int ldc, long sC, int K) {
  __shared__ unsigned short As[2][BM][LP];
  __shared__ unsigned short Bs[2][BN][LP];

  const int z = blockIdx.z;
  A += (size_t)sA * z;
  B += (size_t)sB * z;
  const int bm0 = blockIdx.y * BM, bn0 = blockIdx.x * BN;
  const int tid = threadIdx.x, lane = tid & 31, wid = tid >> 5;
  const int wm = (wid >> 2) * 64, wn = (wid & 3) * 32;

  v8f acc[4][2];
#pragma unroll
  for (int i = 0; i < 4; ++i)
#pragma unroll
    for (int j = 0; j < 2; ++j)
#pragma unroll
      for (int e = 0; e < 8; ++e) acc[i][j][e] = 0.f;

  const int sr = tid >> 1, sc = (tid & 1) * 16;
  auto stage = [&](int buf, int k0) {
    const unsigned short* sa = A + (size_t)(bm0 + sr) * lda + k0 + sc;
    async_ld16(sa,     &As[buf][sr][sc]);
    async_ld16(sa + 8, &As[buf][sr][sc + 8]);
    const unsigned short* sb = B + (size_t)(bn0 + sr) * ldb + k0 + sc;
    async_ld16(sb,     &Bs[buf][sr][sc]);
    async_ld16(sb + 8, &Bs[buf][sr][sc + 8]);
  };

  const int arow = lane & 15, ak = (lane >> 4) * 8;
  const int bcol = lane & 15, bk2 = (lane >> 4) * 8;
  auto compute = [&](int buf) {
    v16bf af[4], bf[2];
#pragma unroll
    for (int i = 0; i < 4; ++i) {   // A-frag: lane=row, K chunks {ak..+7, ak+16..+23}
      const unsigned short* p = &As[buf][wm + i * 16 + arow][ak];
      af[i] = pack16(*(const u16x8*)p, *(const u16x8*)(p + 16));
    }
#pragma unroll
    for (int j = 0; j < 2; ++j) {   // B-frag: lane=col, K chunks {bk2..+7, bk2+16..+23}
      const unsigned short* p = &Bs[buf][wn + j * 16 + bcol][bk2];
      bf[j] = pack16(*(const u16x8*)p, *(const u16x8*)(p + 16));
    }
#pragma unroll
    for (int i = 0; i < 4; ++i)
#pragma unroll
      for (int j = 0; j < 2; ++j) acc[i][j] = wmma_bf16(af[i], bf[j], acc[i][j]);
  };

  const int nk = K / BK;
  stage(0, 0);
  wait_async0();
  __syncthreads();
  for (int t = 0; t < nk - 1; ++t) {        // steady state: branch-free body
    const int buf = t & 1;
    stage(buf ^ 1, (t + 1) * BK);
    compute(buf);
    wait_async0();
    __syncthreads();
  }
  compute((nk - 1) & 1);

  // epilogue: C-frag row = r + 8*(lane>>4), col = lane&15
#pragma unroll
  for (int i = 0; i < 4; ++i)
#pragma unroll
    for (int j = 0; j < 2; ++j) {
      const int col = bn0 + wn + j * 16 + (lane & 15);
      const float bi = bias ? bias[col] : 0.f;
#pragma unroll
      for (int r = 0; r < 8; ++r) {
        const int row = bm0 + wm + i * 16 + r + 8 * (lane >> 4);
        const float v = acc[i][j][r] + bi;
        const size_t off = (size_t)sC * z + (size_t)row * ldc + col;
        if (outF) outF[off] = v;
        if (outB) outB[off] = f2bfu(v);
      }
    }
}

// ---------------------------------------------------------------------------
// Flash attention: one workgroup per (b,h); 8 waves x 32 query rows;
// 32-key blocks; WMMA for Q@K^T and P@V; online softmax per lane-row.
// 65 KB dynamic LDS (CDNA5: 320KB/WGP). K tile staged with async-to-LDS.
// ---------------------------------------------------------------------------
__global__ void __launch_bounds__(256)
attn_kernel(const unsigned short* __restrict__ qkv, unsigned short* __restrict__ o) {
  extern __shared__ char smem[];
  unsigned short* Ks  = (unsigned short*)smem;            // [32][72]  (key, d)
  unsigned short* Vst = (unsigned short*)(smem + 4608);   // [64][40]  (d, key)
  float*          Sws = (float*)(smem + 9728);            // [8][32][33]
  unsigned short* Pws = (unsigned short*)(smem + 43520);  // [8][32][40]
  float*          stat = (float*)(smem + 64000);          // [8][32]

  const int b = blockIdx.x / NH, h = blockIdx.x % NH;
  const int tid = threadIdx.x, lane = tid & 31, wid = tid >> 5;
  const size_t base = (size_t)b * NT * THREE_E;
  const unsigned short* Qp = qkv + base + h * HD;
  const unsigned short* Kp = qkv + base + NE + h * HD;
  const unsigned short* Vp = qkv + base + 2 * NE + h * HD;

  const int qrow0 = wid * 32;

  v16bf aq[2][2];                 // resident Q A-fragments [mtile][ktile]
#pragma unroll
  for (int i = 0; i < 2; ++i)
#pragma unroll
    for (int kk = 0; kk < 2; ++kk) {
      const int r  = qrow0 + i * 16 + (lane & 15);
      const int kb = kk * 32 + (lane >> 4) * 8;
      const unsigned short* p = Qp + (size_t)r * THREE_E + kb;
      aq[i][kk] = pack16(*(const u16x8*)p, *(const u16x8*)(p + 16));
    }

  float mrow = -3.0e38f, lrow = 0.f;
  v8f of[2][4];
#pragma unroll
  for (int i = 0; i < 2; ++i)
#pragma unroll
    for (int j = 0; j < 4; ++j)
#pragma unroll
      for (int e = 0; e < 8; ++e) of[i][j][e] = 0.f;

  for (int kb0 = 0; kb0 < NT; kb0 += 32) {
    // stage K tile [key][d] (async) and V tile transposed [d][key]
    {
      const int key = tid >> 3, db = (tid & 7) * 8;
      async_ld16(Kp + (size_t)(kb0 + key) * THREE_E + db, &Ks[key * 72 + db]);
      u16x8 x = *(const u16x8*)(Vp + (size_t)(kb0 + key) * THREE_E + db);
#pragma unroll
      for (int j = 0; j < 8; ++j) Vst[(db + j) * 40 + key] = x[j];
    }
    wait_async0();
    __syncthreads();

    // S = Q @ K^T  (M=32, N=32 keys, K=64 -> 2 wmma k-steps)
    v8f sf[2][2];
#pragma unroll
    for (int i = 0; i < 2; ++i)
#pragma unroll
      for (int j = 0; j < 2; ++j)
#pragma unroll
        for (int e = 0; e < 8; ++e) sf[i][j][e] = 0.f;
#pragma unroll
    for (int kk = 0; kk < 2; ++kk) {
      v16bf bk[2];
#pragma unroll
      for (int j = 0; j < 2; ++j) {
        const int key = j * 16 + (lane & 15);
        const int d0  = kk * 32 + (lane >> 4) * 8;
        const unsigned short* p = &Ks[key * 72 + d0];
        bk[j] = pack16(*(const u16x8*)p, *(const u16x8*)(p + 16));
      }
#pragma unroll
      for (int i = 0; i < 2; ++i)
#pragma unroll
        for (int j = 0; j < 2; ++j) sf[i][j] = wmma_bf16(aq[i][kk], bk[j], sf[i][j]);
    }
    float* Sw = Sws + wid * 32 * 33;
#pragma unroll
    for (int i = 0; i < 2; ++i)
#pragma unroll
      for (int j = 0; j < 2; ++j)
#pragma unroll
        for (int r = 0; r < 8; ++r)
          Sw[(i * 16 + r + 8 * (lane >> 4)) * 33 + j * 16 + (lane & 15)] = sf[i][j][r];
    __syncthreads();

    // online softmax: lane owns query row (qrow0+lane)
    {
      const float* Srow = Sw + lane * 33;
      float sv[32], mx = -3.0e38f;
#pragma unroll
      for (int c = 0; c < 32; ++c) { sv[c] = Srow[c] * 0.125f; mx = fmaxf(mx, sv[c]); }
      const float mnew = fmaxf(mrow, mx);
      const float corr = __expf(mrow - mnew);
      float ssum = 0.f;
      unsigned short* Prow = Pws + (wid * 32 + lane) * 40;
#pragma unroll
      for (int c = 0; c < 32; ++c) {
        const float p = __expf(sv[c] - mnew);
        ssum += p;
        Prow[c] = f2bfu(p);
      }
      lrow = lrow * corr + ssum;
      mrow = mnew;
      stat[wid * 32 + lane] = corr;
    }
    __syncthreads();

    // rescale O, then O += P @ V  (K=32 keys -> single wmma k-step)
#pragma unroll
    for (int i = 0; i < 2; ++i)
#pragma unroll
      for (int r = 0; r < 8; ++r) {
        const float cf = stat[wid * 32 + i * 16 + r + 8 * (lane >> 4)];
#pragma unroll
        for (int j = 0; j < 4; ++j) of[i][j][r] *= cf;
      }
    v16bf ap[2];
#pragma unroll
    for (int i = 0; i < 2; ++i) {
      const int r  = i * 16 + (lane & 15);
      const int kb = (lane >> 4) * 8;
      const unsigned short* p = Pws + (wid * 32 + r) * 40 + kb;
      ap[i] = pack16(*(const u16x8*)p, *(const u16x8*)(p + 16));
    }
#pragma unroll
    for (int j = 0; j < 4; ++j) {
      const int d  = j * 16 + (lane & 15);
      const int k0 = (lane >> 4) * 8;
      const unsigned short* p = &Vst[d * 40 + k0];
      v16bf bv = pack16(*(const u16x8*)p, *(const u16x8*)(p + 16));
#pragma unroll
      for (int i = 0; i < 2; ++i) of[i][j] = wmma_bf16(ap[i], bv, of[i][j]);
    }
    __syncthreads();
  }

  stat[wid * 32 + lane] = 1.0f / lrow;
  __syncthreads();
#pragma unroll
  for (int i = 0; i < 2; ++i)
#pragma unroll
    for (int j = 0; j < 4; ++j)
#pragma unroll
      for (int r = 0; r < 8; ++r) {
        const int rr = i * 16 + r + 8 * (lane >> 4);
        const float v = of[i][j][r] * stat[wid * 32 + rr];
        const int trow = qrow0 + rr;
        const int d = j * 16 + (lane & 15);
        o[(size_t)(b * NT + trow) * NE + h * HD + d] = f2bfu(v);
      }
}

// ---------------------------------------------------------------------------
// host-side launcher
// ---------------------------------------------------------------------------
extern "C" void kernel_launch(void* const* d_in, const int* in_sizes, int n_in,
                              void* d_out, int out_size, void* d_ws, size_t ws_size,
                              hipStream_t stream) {
  (void)in_sizes; (void)n_in; (void)out_size; (void)ws_size;
  const float* x     = (const float*)d_in[0];
  const float* ln_g  = (const float*)d_in[1];
  const float* ln_b  = (const float*)d_in[2];
  const float* w_in  = (const float*)d_in[3];   // [3,2304,768]
  const float* b_in  = (const float*)d_in[4];   // [6912]
  const float* w_out = (const float*)d_in[5];   // [3,768,768]
  const float* b_out = (const float*)d_in[6];   // [2304]
  const float* U     = (const float*)d_in[7];   // [768,2304]
  const float* Mqkv  = (const float*)d_in[8];   // [6912,2304]

  const long BT = (long)NBATCH * NT;            // 16384
  char* ws = (char*)d_ws;
  size_t off = 0;
  auto carve = [&](size_t bytes) { char* p = ws + off; off = (off + bytes + 255) & ~(size_t)255; return p; };

  unsigned short* xn      = (unsigned short*)carve(BT * NE * 2);
  unsigned short* Ubf     = (unsigned short*)carve((size_t)NE * THREE_E * 2);
  unsigned short* MqkvT   = (unsigned short*)carve((size_t)THREE_E * MK * 2);  // [2304,6912]
  unsigned short* Winbf   = (unsigned short*)carve((size_t)NM * THREE_E * NE * 2);
  unsigned short* Woutbf  = (unsigned short*)carve((size_t)NM * NE * NE * 2);
  unsigned short* UWt     = (unsigned short*)carve((size_t)NE * MK * 2);
  unsigned short* Wfold   = (unsigned short*)carve((size_t)NE * THREE_E * 2);
  unsigned short* WfoldT  = (unsigned short*)carve((size_t)THREE_E * NE * 2);
  unsigned short* Wofold  = (unsigned short*)carve((size_t)NE * THREE_E * 2);
  unsigned short* WofoldT = (unsigned short*)carve((size_t)THREE_E * NE * 2);
  float*          bfold   = (float*)carve((size_t)THREE_E * 4);
  unsigned short* qkvbf   = (unsigned short*)carve(BT * THREE_E * 2);
  unsigned short* obf     = (unsigned short*)carve(BT * NE * 2);

  // 1) conversions + LayerNorm + bias fold
  cvt_f32_bf16<<<2048, 256, 0, stream>>>(U, Ubf, (long)NE * THREE_E);
  cvt_f32_bf16<<<2048, 256, 0, stream>>>(w_in, Winbf, (long)NM * THREE_E * NE);
  cvt_f32_bf16<<<2048, 256, 0, stream>>>(w_out, Woutbf, (long)NM * NE * NE);
  transpose_bf16<<<dim3(THREE_E / 32, MK / 32), 256, 0, stream>>>(
      Mqkv, 1, MqkvT, MK, THREE_E);                       // MqkvT[2304,6912]
  ln_bf16_kernel<<<BT, 256, 0, stream>>>(x, ln_g, ln_b, xn);
  bfold_kernel<<<(THREE_E + 255) / 256, 256, 0, stream>>>(b_in, Mqkv, bfold);

  // 2) weight folding (all B operands stored [N,K])
  // fold1: UWt_m[768,2304] = U_m @ W_m^T
  gemm_bf16_wmma<<<dim3(THREE_E / BN, NE / BM, NM), 256, 0, stream>>>(
      Ubf, THREE_E, (long)NE, Winbf, NE, (long)THREE_E * NE,
      nullptr, nullptr, UWt, MK, (long)THREE_E, NE);
  // fold2: W_fold[768,2304] = UWt[768,6912] @ M_qkv  (B = MqkvT [2304,6912])
  gemm_bf16_wmma<<<dim3(THREE_E / BN, NE / BM, 1), 256, 0, stream>>>(
      UWt, MK, 0L, MqkvT, MK, 0L,
      nullptr, nullptr, Wfold, THREE_E, 0L, MK);
  transpose_bf16<<<dim3(THREE_E / 32, NE / 32), 256, 0, stream>>>(
      Wfold, 0, WfoldT, NE, THREE_E);                     // WfoldT[2304,768]
  // ofold: W_ofold_m[768,768] = U_m @ Wout_m^T
  gemm_bf16_wmma<<<dim3(NE / BN, NE / BM, NM), 256, 0, stream>>>(
      Ubf, THREE_E, (long)NE, Woutbf, NE, (long)NE * NE,
      nullptr, nullptr, Wofold, THREE_E, (long)NE, NE);
  transpose_bf16<<<dim3(THREE_E / 32, NE / 32), 256, 0, stream>>>(
      Wofold, 0, WofoldT, NE, THREE_E);                   // WofoldT[2304,768]

  // 3) qkv = xn @ W_fold + b_fold
  gemm_bf16_wmma<<<dim3(THREE_E / BN, BT / BM, 1), 256, 0, stream>>>(
      xn, NE, 0L, WfoldT, NE, 0L,
      bfold, nullptr, qkvbf, THREE_E, 0L, NE);

  // 4) attention (65 KB dynamic LDS; CDNA5 allows 320 KB/WGP)
  attn_kernel<<<NBATCH * NH, 256, 65024, stream>>>(qkvbf, obf);

  // 5) out = o @ W_ofold + out_bias -> fp32 d_out
  gemm_bf16_wmma<<<dim3(THREE_E / BN, BT / BM, 1), 256, 0, stream>>>(
      obf, NE, 0L, WofoldT, NE, 0L,
      b_out, (float*)d_out, nullptr, THREE_E, 0L, NE);
}
